// Flow_15753940042285
// MI455X (gfx1250) — matrix operand order repaired and checked by reading, not compiled, not run on hardware
//
#include <hip/hip_runtime.h>

// ---------------------------------------------------------------------------
// Flow-matching MLP forward for MI455X (gfx1250), wave32 + WMMA f16->f32.
//
// Pass 1 (pack_weights): convert all fp32 weight matrices to f16 in exact
//   WMMA B-fragment order (zero K-padding baked in) into d_ws (164 KB).
// Pass 2 (flow_fwd): one wave32 per 16-row tile; every matmul is
//   v_wmma_f32_16x16x32_f16. A-fragments come from f16 activation tiles in
//   per-wave LDS (ds_load_b128); B-fragments are coalesced 32B/lane
//   global_load_b128 pairs from the packed L2-resident weights.
// N-tile loops capped at unroll 4 to keep live VGPRs < 256 (no vgpr_msb churn).
// ---------------------------------------------------------------------------

typedef _Float16 h16;
typedef __attribute__((ext_vector_type(16))) _Float16 v16h;
typedef __attribute__((ext_vector_type(8)))  _Float16 v8h;
typedef __attribute__((ext_vector_type(8)))  float    v8f;

#define WAVES_PER_BLOCK 2
#define ROWS_PER_WAVE   16
#define ROWS_PER_BLOCK  (WAVES_PER_BLOCK * ROWS_PER_WAVE)

// Per-wave LDS layout (in halves); all offsets 32B-aligned.
#define OFF_BUF0 0        // 16 x 96 (block1 input; later 16x64 y stage)
#define OFF_SCR  1536     // 16 x 64
#define OFF_H    2560     // 16 x 64
#define OFF_Y    3584     // 16 x 64
#define OFF_QKV  4608     // 2 x 16 x 192
#define OFF_O    10752    // 16 x 64
#define WAVE_HALVES 11776

// input indices in setup_inputs() order
enum {
  IX = 0, IT, IY,
  ILP_W1, ILP_B1, ILP_W2, ILP_B2,
  IA1_WIN, IA1_BIN, IA1_WOUT, IA1_BOUT,
  IA2_WIN, IA2_BIN, IA2_WOUT, IA2_BOUT,
  IF1_GW, IF1_GB, IF1_BW, IF1_BB,
  IF2_GW, IF2_GB, IF2_BW, IF2_BB,
  IB1_W1, IB1_B1, IB1_W2, IB1_B2,
  IB2_W1, IB2_B1, IB2_W2, IB2_B2,
  IB3_W1, IB3_B1, IB3_W2, IB3_B2,
  IOUT_W, IOUT_B, N_IN
};

// Fragment-base offsets (units of fragments; 1 fragment = 32x16 f16 = 512 halves)
enum {
  FO_LP_W1  = 0,    // 40x32  pad64  NT=2  NK=2 -> 4
  FO_LP_W2  = 4,    // 32x64         NT=4  NK=1 -> 4
  FO_A1_WIN = 8,    // 64x192        NT=12 NK=2 -> 24
  FO_A1_WOUT= 32,   // 64x64         NT=4  NK=2 -> 8
  FO_A2_WIN = 40,
  FO_A2_WOUT= 64,
  FO_F1_GW  = 72,
  FO_F1_BW  = 80,
  FO_F2_GW  = 88,
  FO_F2_BW  = 96,
  FO_B1_W1  = 104,  // 80x64 pad96   NT=4  NK=3 -> 12
  FO_B1_W2  = 116,
  FO_B2_W1  = 124,
  FO_B2_W2  = 132,
  FO_B3_W1  = 140,
  FO_B3_W2  = 148,
  FO_OUT_W  = 156,
  N_FRAGS   = 164   // 164 KB in d_ws
};

#define N_MATS 17

struct PackDesc { int src; int K; int NT; int NK; int off; };
struct PackParams { const float* srcs[N_IN]; h16* dst; PackDesc d[N_MATS]; };

// --------------------------- pass 1: weight packer --------------------------
__global__ __launch_bounds__(32) void pack_weights(PackParams P) {
  PackDesc D = P.d[blockIdx.x];
  const float* w = P.srcs[D.src];
  const int lane = (int)(threadIdx.x & 31u);
  const int N = D.NT * 16;
  const int nfrag = D.NT * D.NK;
  const int n_in_frag = lane & 15;
  const int kb_in_frag = (lane >> 4) << 3;     // lanes 0-15: K 0..7,16..23; 16-31: K 8..15,24..31
  for (int f = 0; f < nfrag; ++f) {
    int nt = f / D.NK, kc = f - nt * D.NK;
    int n  = nt * 16 + n_in_frag;
    int kb = kc * 32 + kb_in_frag;
    h16* out = P.dst + (size_t)(D.off + f) * 512 + lane * 16;
#pragma unroll
    for (int i = 0; i < 16; ++i) {
      int k = kb + i + (i >= 8 ? 8 : 0);
      float v = (k < D.K) ? w[k * N + n] : 0.0f;  // zero K-padding baked in
      out[i] = (h16)v;
    }
  }
}

// ------------------------------- pass 2 helpers -----------------------------
__device__ __forceinline__ int laneid() { return (int)(threadIdx.x & 31u); }

// A fragment 16x32 f16 from LDS (two 16B vector loads -> ds_load_b128)
__device__ __forceinline__ v16h load_A(const h16* p0, int stride, int k0) {
  int lane = laneid();
  int m  = lane & 15;
  int kb = (lane >> 4) << 3;
  const h16* p = p0 + m * stride + k0 + kb;
  v8h lo = *(const v8h*)p;
  v8h hi = *(const v8h*)(p + 16);
  v16h a;
#pragma unroll
  for (int i = 0; i < 8; ++i) { a[i] = lo[i]; a[8 + i] = hi[i]; }
  return a;
}

// B fragment: one contiguous 32B/lane read from packed weights (coalesced)
__device__ __forceinline__ v16h load_Bp(const h16* wp, int frag) {
  return *(const v16h*)(wp + (size_t)((frag << 5) + laneid()) * 16);
}

__device__ __forceinline__ v8f bias_C(const float* bias, int n0) {
  float bv = bias[n0 + (laneid() & 15)];
  v8f c;
#pragma unroll
  for (int r = 0; r < 8; ++r) c[r] = bv;
  return c;
}

__device__ __forceinline__ v8f wmma16(v16h a, v16h b, v8f c) {
  return __builtin_amdgcn_wmma_f32_16x16x32_f16(false, a, false, b, (short)0, c,
                                                false, false);
}

__device__ __forceinline__ void store_D(h16* d, int stride, int n0, v8f acc, bool relu) {
  int lane = laneid();
  int n  = n0 + (lane & 15);
  int mb = (lane >> 4) << 3;
#pragma unroll
  for (int r = 0; r < 8; ++r) {
    float v = acc[r];
    if (relu) v = fmaxf(v, 0.0f);
    d[(mb + r) * stride + n] = (h16)v;
  }
}

// D[16xN] = act( A[16x(32*NK)] @ Wpacked + bias ); unroll capped at 4 N-tiles
template <int NK, int N>
__device__ __forceinline__ void gemm_p(const h16* A, int As,
                                       const h16* wp, const float* bias,
                                       h16* D, int Ds, bool relu) {
  v16h a[NK];
#pragma unroll
  for (int c = 0; c < NK; ++c) a[c] = load_A(A, As, c * 32);
#pragma unroll 4
  for (int n0 = 0; n0 < N; n0 += 16) {
    v8f acc = bias_C(bias, n0);
#pragma unroll
    for (int c = 0; c < NK; ++c)
      acc = wmma16(a[c], load_Bp(wp, (n0 >> 4) * NK + c), acc);
    store_D(D, Ds, n0, acc, relu);
  }
}

// h += Lin2(ReLU(Lin1(h)))
__device__ __forceinline__ void res_block(h16* hbuf, h16* scr,
                                          const h16* w1p, const float* b1,
                                          const h16* w2p, const float* b2) {
  gemm_p<2, 64>(hbuf, 64, w1p, b1, scr, 64, true);
  v16h a0 = load_A(scr, 64, 0);
  v16h a1 = load_A(scr, 64, 32);
  int lane = laneid();
  int nl = lane & 15;
  int mb = (lane >> 4) << 3;
#pragma unroll 4
  for (int n0 = 0; n0 < 64; n0 += 16) {
    v8f acc = bias_C(b2, n0);
    acc = wmma16(a0, load_Bp(w2p, (n0 >> 4) * 2 + 0), acc);
    acc = wmma16(a1, load_Bp(w2p, (n0 >> 4) * 2 + 1), acc);
#pragma unroll
    for (int r = 0; r < 8; ++r)
      acc[r] += (float)hbuf[(mb + r) * 64 + n0 + nl];
#pragma unroll
    for (int r = 0; r < 8; ++r)
      hbuf[(mb + r) * 64 + n0 + nl] = (h16)acc[r];
  }
}

// 2-token MHA (token0 = h, token1 = y_embed) + FiLM, fused
__device__ __forceinline__ void mha_film(h16* hbuf, h16* qkv, h16* obuf,
                                         v16h ay0, v16h ay1,
                                         const h16* winp, const float* bin,
                                         const h16* woutp, const float* bout,
                                         const h16* gwp, const float* gb,
                                         const h16* bwp, const float* bb) {
  gemm_p<2, 192>(hbuf, 64, winp, bin, qkv, 192, false);      // token 0
#pragma unroll 4
  for (int n0 = 0; n0 < 192; n0 += 16) {                     // token 1 (y)
    v8f acc = bias_C(bin, n0);
    acc = wmma16(ay0, load_Bp(winp, (n0 >> 4) * 2 + 0), acc);
    acc = wmma16(ay1, load_Bp(winp, (n0 >> 4) * 2 + 1), acc);
    store_D(qkv + 16 * 192, 192, n0, acc, false);
  }

  // attention: one lane per (row, head); 64 tasks over 2 iterations
  int lane = laneid();
#pragma unroll
  for (int it = 0; it < 2; ++it) {
    int task = lane + it * 32;
    int row  = task & 15;
    int head = task >> 4;                                    // 0..3, D = 16
    const h16* base = qkv + row * 192 + head * 16;
    v16h qv = *(const v16h*)(base);
    v16h k0 = *(const v16h*)(base + 64);
    v16h k1 = *(const v16h*)(base + 64 + 16 * 192);
    v16h v0 = *(const v16h*)(base + 128);
    v16h v1 = *(const v16h*)(base + 128 + 16 * 192);
    float s0 = 0.0f, s1 = 0.0f;
#pragma unroll
    for (int j = 0; j < 16; ++j) {
      float qf = (float)qv[j];
      s0 += qf * (float)k0[j];
      s1 += qf * (float)k1[j];
    }
    s0 *= 0.25f; s1 *= 0.25f;                                // 1/sqrt(16)
    float mx = fmaxf(s0, s1);
    float e0 = __expf(s0 - mx), e1 = __expf(s1 - mx);
    float inv = 1.0f / (e0 + e1);
    float w0 = e0 * inv, w1 = e1 * inv;
    v16h o;
#pragma unroll
    for (int j = 0; j < 16; ++j)
      o[j] = (h16)(w0 * (float)v0[j] + w1 * (float)v1[j]);
    *(v16h*)(obuf + row * 64 + head * 16) = o;
  }

  // out-projection + FiLM per N-tile: h = (y@gw+gb) * (o@wout+bout) + (y@bw+bb)
  v16h ao0 = load_A(obuf, 64, 0);
  v16h ao1 = load_A(obuf, 64, 32);
#pragma unroll 2
  for (int n0 = 0; n0 < 64; n0 += 16) {
    int nt2 = (n0 >> 4) * 2;
    v8f att = bias_C(bout, n0);
    att = wmma16(ao0, load_Bp(woutp, nt2 + 0), att);
    att = wmma16(ao1, load_Bp(woutp, nt2 + 1), att);
    v8f gam = bias_C(gb, n0);
    gam = wmma16(ay0, load_Bp(gwp, nt2 + 0), gam);
    gam = wmma16(ay1, load_Bp(gwp, nt2 + 1), gam);
    v8f bet = bias_C(bb, n0);
    bet = wmma16(ay0, load_Bp(bwp, nt2 + 0), bet);
    bet = wmma16(ay1, load_Bp(bwp, nt2 + 1), bet);
    v8f hv;
#pragma unroll
    for (int r = 0; r < 8; ++r) hv[r] = gam[r] * att[r] + bet[r];
    store_D(hbuf, 64, n0, hv, false);
  }
}

struct Params {
  const float* p[N_IN];
  const h16* wf;        // packed weight fragments (d_ws)
  float* out;
  int nrows;
};

__global__ __launch_bounds__(WAVES_PER_BLOCK * 32) void flow_fwd(Params P) {
  __shared__ __align__(128) h16 smem[WAVES_PER_BLOCK * WAVE_HALVES];
  const int wave = (int)(threadIdx.x >> 5);
  const int lane = laneid();
  h16* ws   = smem + wave * WAVE_HALVES;
  h16* buf0 = ws + OFF_BUF0;
  h16* scr  = ws + OFF_SCR;
  h16* hbuf = ws + OFF_H;
  h16* ybuf = ws + OFF_Y;
  h16* qkv  = ws + OFF_QKV;
  h16* obuf = ws + OFF_O;
  const h16* W = P.wf;

  const int row0 = ((int)blockIdx.x * WAVES_PER_BLOCK + wave) * ROWS_PER_WAVE;
  const int last = P.nrows - 1;

  // ---- stage x into buf0[.., 0..63] (stride 96), coalesced ----
#pragma unroll 8
  for (int i = 0; i < 32; ++i) {
    int idx = i * 32 + lane;                    // 16*64 = 1024
    int r = idx >> 6, c = idx & 63;
    int gr = row0 + r; if (gr > last) gr = last;
    buf0[r * 96 + c] = (h16)P.p[IX][gr * 64 + c];
  }
  // ---- time features: cols 64..71 = sin(t*2^i*pi), 72..79 = cos ----
  if (lane < 16) {
    int gr = row0 + lane; if (gr > last) gr = last;
    float arg = P.p[IT][gr] * 3.14159265358979323846f;
#pragma unroll
    for (int i = 0; i < 8; ++i) {
      buf0[lane * 96 + 64 + i] = (h16)__sinf(arg);
      buf0[lane * 96 + 72 + i] = (h16)__cosf(arg);
      arg *= 2.0f;
    }
  }
  // ---- zero K-padding cols 80..95 ----
#pragma unroll
  for (int i = 0; i < 8; ++i) {
    int idx = i * 32 + lane;                    // 16*16 = 256
    int r = idx >> 4, c = 80 + (idx & 15);
    buf0[r * 96 + c] = (h16)0.0f;
  }

  // ---- block1: 80->64 (K pad 96) -> ReLU -> 64->64 ----
  gemm_p<3, 64>(buf0, 96, W + FO_B1_W1 * 512, P.p[IB1_B1], scr, 64, true);
  gemm_p<2, 64>(scr, 64, W + FO_B1_W2 * 512, P.p[IB1_B2], hbuf, 64, false);

  // ---- stage y (16x40, unconditional) then zero-pad cols 40..63 ----
#pragma unroll 5
  for (int i = 0; i < 20; ++i) {
    int idx = i * 32 + lane;                    // 16*40 = 640
    int r = idx / 40, c = idx - r * 40;
    int gr = row0 + r; if (gr > last) gr = last;
    buf0[r * 64 + c] = (h16)P.p[IY][gr * 40 + c];
  }
#pragma unroll
  for (int i = 0; i < 12; ++i) {
    int idx = i * 32 + lane;                    // 16*24 = 384
    int r = idx / 24, c = 40 + (idx - r * 24);
    buf0[r * 64 + c] = (h16)0.0f;
  }
  gemm_p<2, 32>(buf0, 64, W + FO_LP_W1 * 512, P.p[ILP_B1], scr, 32, true);
  gemm_p<1, 64>(scr, 32, W + FO_LP_W2 * 512, P.p[ILP_B2], ybuf, 64, false);

  // y_embed A-fragments stay in registers for both MHA + FiLM layers
  v16h ay0 = load_A(ybuf, 64, 0);
  v16h ay1 = load_A(ybuf, 64, 32);

  mha_film(hbuf, qkv, obuf, ay0, ay1,
           W + FO_A1_WIN * 512, P.p[IA1_BIN], W + FO_A1_WOUT * 512, P.p[IA1_BOUT],
           W + FO_F1_GW * 512, P.p[IF1_GB], W + FO_F1_BW * 512, P.p[IF1_BB]);
  res_block(hbuf, scr, W + FO_B2_W1 * 512, P.p[IB2_B1], W + FO_B2_W2 * 512, P.p[IB2_B2]);

  mha_film(hbuf, qkv, obuf, ay0, ay1,
           W + FO_A2_WIN * 512, P.p[IA2_BIN], W + FO_A2_WOUT * 512, P.p[IA2_BOUT],
           W + FO_F2_GW * 512, P.p[IF2_GB], W + FO_F2_BW * 512, P.p[IF2_BB]);
  res_block(hbuf, scr, W + FO_B3_W1 * 512, P.p[IB3_B1], W + FO_B3_W2 * 512, P.p[IB3_B2]);

  // ---- output projection straight to global (fp32) ----
  v16h ah0 = load_A(hbuf, 64, 0);
  v16h ah1 = load_A(hbuf, 64, 32);
  int nl = lane & 15;
  int mb = (lane >> 4) << 3;
#pragma unroll 4
  for (int n0 = 0; n0 < 64; n0 += 16) {
    v8f acc = bias_C(P.p[IOUT_B], n0);
    acc = wmma16(ah0, load_Bp(W + FO_OUT_W * 512, (n0 >> 4) * 2 + 0), acc);
    acc = wmma16(ah1, load_Bp(W + FO_OUT_W * 512, (n0 >> 4) * 2 + 1), acc);
#pragma unroll
    for (int r = 0; r < 8; ++r) {
      int gr = row0 + mb + r;
      if (gr < P.nrows) P.out[gr * 64 + n0 + nl] = acc[r];
    }
  }
}

extern "C" void kernel_launch(void* const* d_in, const int* in_sizes, int n_in,
                              void* d_out, int out_size, void* d_ws, size_t ws_size,
                              hipStream_t stream) {
  (void)ws_size; (void)out_size;

  // pass 1: pack weights into d_ws as f16 B-fragments
  PackParams PP;
  for (int i = 0; i < N_IN && i < n_in; ++i) PP.srcs[i] = (const float*)d_in[i];
  PP.dst = (h16*)d_ws;
  const PackDesc descs[N_MATS] = {
    {ILP_W1,  40,  2, 2, FO_LP_W1},
    {ILP_W2,  32,  4, 1, FO_LP_W2},
    {IA1_WIN, 64, 12, 2, FO_A1_WIN},
    {IA1_WOUT,64,  4, 2, FO_A1_WOUT},
    {IA2_WIN, 64, 12, 2, FO_A2_WIN},
    {IA2_WOUT,64,  4, 2, FO_A2_WOUT},
    {IF1_GW,  64,  4, 2, FO_F1_GW},
    {IF1_BW,  64,  4, 2, FO_F1_BW},
    {IF2_GW,  64,  4, 2, FO_F2_GW},
    {IF2_BW,  64,  4, 2, FO_F2_BW},
    {IB1_W1,  80,  4, 3, FO_B1_W1},
    {IB1_W2,  64,  4, 2, FO_B1_W2},
    {IB2_W1,  64,  4, 2, FO_B2_W1},
    {IB2_W2,  64,  4, 2, FO_B2_W2},
    {IB3_W1,  64,  4, 2, FO_B3_W1},
    {IB3_W2,  64,  4, 2, FO_B3_W2},
    {IOUT_W,  64,  4, 2, FO_OUT_W},
  };
  for (int i = 0; i < N_MATS; ++i) PP.d[i] = descs[i];
  pack_weights<<<N_MATS, 32, 0, stream>>>(PP);

  // pass 2: fused forward
  Params P;
  for (int i = 0; i < N_IN && i < n_in; ++i) P.p[i] = (const float*)d_in[i];
  P.wf    = (const h16*)d_ws;
  P.out   = (float*)d_out;
  P.nrows = in_sizes[0] / 64;                  // x is [B, 64]

  int nblocks = (P.nrows + ROWS_PER_BLOCK - 1) / ROWS_PER_BLOCK;
  flow_fwd<<<nblocks, WAVES_PER_BLOCK * 32, 0, stream>>>(P);
}
